// GraphAttentionLayer_72121090834944
// MI455X (gfx1250) — matrix operand order repaired and checked by reading, not compiled, not run on hardware
//
#include <hip/hip_runtime.h>
#include <hip/hip_bf16.h>
#include <stdint.h>

// ---- problem constants (fixed by the reference) ----
#define N_NODES 8192
#define F_IN    512
#define F_OUT   64
#define LRELU_ALPHA 0.2f
#define NEG_INF_F (-9.0e15f)

typedef __attribute__((ext_vector_type(2)))  float  v2f;
typedef __attribute__((ext_vector_type(8)))  float  v8f;
typedef __attribute__((ext_vector_type(2)))  __bf16 v2bf;
typedef __attribute__((ext_vector_type(16))) __bf16 v16bf;

// Pack two f32 into one DWORD of bf16x2 via HW v_cvt_pk_bf16_f32.
__device__ __forceinline__ unsigned pack_bf16(float lo, float hi) {
    v2f f;
    f.x = lo;
    f.y = hi;
    v2bf b = __builtin_convertvector(f, v2bf);
    return __builtin_bit_cast(unsigned, b);
}

// ---------------------------------------------------------------------------
// Kernel A: h = x @ W (8192x512 @ 512x64, f32), f32 WMMA 16x16x4.
// Outputs h transposed in BOTH precisions:
//   hTf [F_OUT][N] f32  -> coalesced reads in the src/dst kernel
//   hTb [F_OUT][N] bf16 -> contiguous 16B B-operand K-runs for the flash kernel
// C/D fragment rows are contiguous along M, so all stores are b128.
// ---------------------------------------------------------------------------
__global__ __launch_bounds__(32)
void gat_proj_kernel(const float* __restrict__ x, const float* __restrict__ W,
                     float* __restrict__ hTf, unsigned short* __restrict__ hTb) {
    const int lane = threadIdx.x;
    const int r    = lane & 15;
    const int hf   = lane >> 4;            // which half of the wave
    const int i0   = blockIdx.x * 16;

    v8f acc[4] = {};
    const float* xrow = x + (size_t)(i0 + r) * F_IN;
    for (int k = 0; k < F_IN; k += 4) {
        // A 16x4 f32: VGPR0 = K0|K2, VGPR1 = K1|K3  -> float2 per lane
        v2f A = *(const v2f*)(xrow + k + 2 * hf);
#pragma unroll
        for (int nt = 0; nt < 4; ++nt) {
            v2f B;
            B.x = W[(size_t)(k + 2 * hf    ) * F_OUT + nt * 16 + r];
            B.y = W[(size_t)(k + 2 * hf + 1) * F_OUT + nt * 16 + r];
            acc[nt] = __builtin_amdgcn_wmma_f32_16x16x4_f32(
                false, A, false, B, (short)0, acc[nt], false, false);
        }
    }
#pragma unroll
    for (int nt = 0; nt < 4; ++nt) {
        const int col = nt * 16 + r;
        // rows i0+8*hf .. +7 are contiguous in the transposed layouts
        union { float4 q[2]; float f[8]; } fs;
#pragma unroll
        for (int rr = 0; rr < 8; ++rr) fs.f[rr] = acc[nt][rr];
        float4* fp = (float4*)(hTf + (size_t)col * N_NODES + i0 + 8 * hf);
        fp[0] = fs.q[0];
        fp[1] = fs.q[1];

        union { uint4 q; unsigned u[4]; } pk;
#pragma unroll
        for (int p = 0; p < 4; ++p)
            pk.u[p] = pack_bf16(acc[nt][2 * p], acc[nt][2 * p + 1]);
        *(uint4*)(hTb + (size_t)col * N_NODES + i0 + 8 * hf) = pk.q;
    }
}

// ---------------------------------------------------------------------------
// Kernel C: src_i = h_i . a[:F], dst_i = h_i . a[F:]
// Reads hTf[f][i]: consecutive threads hit consecutive addresses (coalesced),
// a[f] is wave-uniform (scalar loads).
// ---------------------------------------------------------------------------
__global__ void gat_attvec_kernel(const float* __restrict__ hTf, const float* __restrict__ a,
                                  float* __restrict__ src, float* __restrict__ dst) {
    const int i = blockIdx.x * blockDim.x + threadIdx.x;
    if (i >= N_NODES) return;
    float s = 0.f, d = 0.f;
#pragma unroll 8
    for (int f = 0; f < F_OUT; ++f) {
        const float v = hTf[(size_t)f * N_NODES + i];
        s += v * a[f];
        d += v * a[F_OUT + f];
    }
    src[i] = s;
    dst[i] = d;
}

// ---------------------------------------------------------------------------
// Kernel B: fused masked-softmax + aggregation (flash style).
// One wave per 16 output rows; streams adj exactly once (~256MB, the HBM
// roofline term), online softmax stats in registers, P.H via bf16 WMMA
// 16x16x32. Accumulator rescale is skipped with a wave-uniform __ballot
// branch when no row's running max grew (the common case once the max
// stabilizes) -- EXEC stays all-ones, as WMMA requires.
// ---------------------------------------------------------------------------
__global__ __launch_bounds__(32)
void gat_softmax_agg_kernel(const int* __restrict__ adj,
                            const unsigned short* __restrict__ hTb,
                            const float* __restrict__ src,
                            const float* __restrict__ dst,
                            float* __restrict__ out) {
    const int lane = threadIdx.x;
    const int r    = lane & 15;
    const int hf   = lane >> 4;
    const int i0   = blockIdx.x * 16;
    const int row  = i0 + r;

    const float sv = src[row];
    float m = -3.0e38f;      // running row max
    float l = 0.0f;          // running row sum
    v8f acc[4] = {};

    const int* __restrict__ arow = adj + (size_t)row * N_NODES;

    for (int j0 = 0; j0 < N_NODES; j0 += 32) {
        const int jb = j0 + 8 * hf;   // this lane's first j (A-operand K layout)

        // prefetch next chunk of this adj row (global_prefetch_b8)
        if (j0 + 32 < N_NODES) __builtin_prefetch((const void*)(arow + jb + 32), 0, 0);

        const float4 d0 = *(const float4*)(dst + jb);
        const float4 d1 = *(const float4*)(dst + jb + 4);
        const float4 d2 = *(const float4*)(dst + jb + 16);
        const float4 d3 = *(const float4*)(dst + jb + 20);
        const int4 q0 = *(const int4*)(arow + jb);
        const int4 q1 = *(const int4*)(arow + jb + 4);
        const int4 q2 = *(const int4*)(arow + jb + 16);
        const int4 q3 = *(const int4*)(arow + jb + 20);

        const float dv[16] = {d0.x,d0.y,d0.z,d0.w, d1.x,d1.y,d1.z,d1.w,
                              d2.x,d2.y,d2.z,d2.w, d3.x,d3.y,d3.z,d3.w};
        const int   av[16] = {q0.x,q0.y,q0.z,q0.w, q1.x,q1.y,q1.z,q1.w,
                              q2.x,q2.y,q2.z,q2.w, q3.x,q3.y,q3.z,q3.w};

        float e[16];
        float cmax = -3.0e38f;
#pragma unroll
        for (int t = 0; t < 16; ++t) {
            float ev = sv + dv[t];
            ev = fmaxf(ev, LRELU_ALPHA * ev);        // leaky relu
            ev = (av[t] > 0) ? ev : NEG_INF_F;       // adjacency mask
            e[t] = ev;
            cmax = fmaxf(cmax, ev);
        }
        // combine max across the two half-waves handling the same row
        cmax = fmaxf(cmax, __shfl_xor(cmax, 16));
        const float m_new = fmaxf(m, cmax);

        // Rescale only if some row's max actually grew (wave-uniform branch).
        if (__ballot(m_new > m)) {
            const float scale = __expf(m - m_new);
            l *= scale;
            // acc VGPR rr holds row rr+8*hf; its scale lives in lane rr+8*hf
#pragma unroll
            for (int rr = 0; rr < 8; ++rr) {
                const float sc = __shfl(scale, rr + 8 * hf);
#pragma unroll
                for (int nt = 0; nt < 4; ++nt) acc[nt][rr] *= sc;
            }
        }
        m = m_new;

        float psum = 0.f;
        union { v16bf v; unsigned u[8]; } A;
#pragma unroll
        for (int v = 0; v < 8; ++v) {                // A VGPR v = K pair (2v,2v+1)
            const float p0 = __expf(e[2 * v]     - m_new);
            const float p1 = __expf(e[2 * v + 1] - m_new);
            psum += p0 + p1;
            A.u[v] = pack_bf16(p0, p1);              // v_cvt_pk_bf16_f32
        }
        l += psum + __shfl_xor(psum, 16);

        // acc += P(16x32,bf16) x H(32x16,bf16) for each of 4 N-tiles
#pragma unroll
        for (int nt = 0; nt < 4; ++nt) {
            union { v16bf v; uint4 q[2]; } B;
            const uint4* bp = (const uint4*)(hTb + (size_t)(nt * 16 + r) * N_NODES
                                                 + j0 + 8 * hf);
            B.q[0] = bp[0];
            B.q[1] = bp[2];
            acc[nt] = __builtin_amdgcn_wmma_f32_16x16x32_bf16(
                false, A.v, false, B.v, (short)0, acc[nt], false, false);
        }
    }

    // finalize: divide by row sum, ELU, store
#pragma unroll
    for (int rr = 0; rr < 8; ++rr) {
        const float lr  = __shfl(l, rr + 8 * hf);
        const float inv = 1.0f / lr;
        const int orow  = i0 + rr + 8 * hf;
#pragma unroll
        for (int nt = 0; nt < 4; ++nt) {
            float v = acc[nt][rr] * inv;
            v = (v > 0.f) ? v : (__expf(v) - 1.0f);  // ELU(alpha=1)
            out[(size_t)orow * F_OUT + nt * 16 + r] = v;
        }
    }
}

// ---------------------------------------------------------------------------
extern "C" void kernel_launch(void* const* d_in, const int* in_sizes, int n_in,
                              void* d_out, int out_size, void* d_ws, size_t ws_size,
                              hipStream_t stream) {
    const float* x   = (const float*)d_in[0];   // [N, F_IN]
    const int*   adj = (const int*)  d_in[1];   // [N, N]
    const float* W   = (const float*)d_in[2];   // [F_IN, F_OUT]
    const float* a   = (const float*)d_in[3];   // [2*F_OUT]
    float* out = (float*)d_out;                 // [N, F_OUT]

    // workspace: hTf f32 [64][8192] (2MB) | hTb bf16 [64][8192] (1MB) | src | dst
    float* hTf = (float*)d_ws;
    unsigned short* hTb = (unsigned short*)(hTf + (size_t)F_OUT * N_NODES);
    float* src = (float*)(hTb + (size_t)F_OUT * N_NODES);
    float* dst = src + N_NODES;

    gat_proj_kernel<<<N_NODES / 16, 32, 0, stream>>>(x, W, hTf, hTb);
    gat_attvec_kernel<<<N_NODES / 256, 256, 0, stream>>>(hTf, a, src, dst);
    gat_softmax_agg_kernel<<<N_NODES / 16, 32, 0, stream>>>(adj, hTb, src, dst, out);
}
